// ExecutionSystem_16793322127878
// MI455X (gfx1250) — matrix operand order repaired and checked by reading, not compiled, not run on hardware
//
#include <hip/hip_runtime.h>
#include <hip/hip_bf16.h>
#include <stdint.h>

typedef __attribute__((ext_vector_type(16))) __bf16 v16bf;
typedef __attribute__((ext_vector_type(8)))  float  v8f;

#define NT    512
#define NWAVE 16

#define Bv  8
#define Sv  1024
#define Dv  512
#define Hv  256
#define Nv  64
#define SDv 128
#define NHv 4
#define DHv 64

struct bf16x4 { __bf16 a, b, c, d; };

__device__ __forceinline__ void store_bf4(__bf16* p, float4 v) {
  bf16x4 t; t.a = (__bf16)v.x; t.b = (__bf16)v.y; t.c = (__bf16)v.z; t.d = (__bf16)v.w;
  *(bf16x4*)p = t;
}
__device__ __forceinline__ float4 f4add(float4 a, float4 b) {
  return make_float4(a.x + b.x, a.y + b.y, a.z + b.z, a.w + b.w);
}
__device__ __forceinline__ float dot4(float4 a, float4 b) {
  return fmaf(a.x, b.x, fmaf(a.y, b.y, fmaf(a.z, b.z, a.w * b.w)));
}
__device__ __forceinline__ float wave_sum(float v) {
#pragma unroll
  for (int m = 16; m; m >>= 1) v += __shfl_xor(v, m, 32);
  return v;
}
__device__ __forceinline__ float wave_max(float v) {
#pragma unroll
  for (int m = 16; m; m >>= 1) v = fmaxf(v, __shfl_xor(v, m, 32));
  return v;
}
__device__ __forceinline__ float grp16_sum(float v) {
#pragma unroll
  for (int m = 8; m; m >>= 1) v += __shfl_xor(v, m, 32);
  return v;
}
__device__ __forceinline__ float sigmoidf_(float x) { return 1.f / (1.f + __expf(-x)); }

// ---------------- WMMA fragment helpers (gfx1250 wave32) ----------------
// A 16x32 bf16 tile from LDS row-major [16][K]:
//  lane L (row L%16): L<16 -> elems 0..7 = K 0..7, elems 8..15 = K 16..23
//                     L>=16 -> elems 0..7 = K 8..15, elems 8..15 = K 24..31
__device__ __forceinline__ v16bf load_a_frag(const __bf16* A, int ldk, int kc, int lane) {
  const __bf16* p = A + (lane & 15) * ldk + kc * 32 + ((lane & 16) ? 8 : 0);
  v16bf r;
#pragma unroll
  for (int i = 0; i < 8; ++i) { r[i] = p[i]; r[i + 8] = p[16 + i]; }
  return r;
}
// B 32x16 bf16 tile: pre-packed so each lane reads 16 contiguous bf16 (32B).
__device__ __forceinline__ v16bf load_b_frag(const __bf16* pack, int nkc, int tile, int kc, int lane) {
  const __bf16* p = pack + ((((size_t)tile * nkc + kc) * 32) + lane) * 16;
  v16bf r;
#pragma unroll
  for (int i = 0; i < 16; ++i) r[i] = p[i];
  return r;
}

// Workgroup GEMM: gof[16][N] (LDS f32) = A(16xK LDS bf16) * packedW(KxN), K-loop double-buffered.
__device__ __forceinline__ void gemm_ws(const __bf16* A, int K, const __bf16* pack, int N,
                                        float* gof, int wave, int lane) {
  const int nkc = K >> 5, ntile = N >> 4;
  for (int tile = wave; tile < ntile; tile += NWAVE) {
    v8f acc = {};
    v16bf a = load_a_frag(A, K, 0, lane);
    v16bf b = load_b_frag(pack, nkc, tile, 0, lane);
    for (int kc = 0; kc < nkc; ++kc) {
      const int kn = (kc + 1 < nkc) ? kc + 1 : kc;
      v16bf an = load_a_frag(A, K, kn, lane);
      v16bf bn = load_b_frag(pack, nkc, tile, kn, lane);
      acc = __builtin_amdgcn_wmma_f32_16x16x32_bf16(false, a, false, b, (short)0, acc, false, false);
      a = an; b = bn;
    }
    const int col = tile * 16 + (lane & 15);
    const int mb  = (lane & 16) ? 8 : 0;
#pragma unroll
    for (int i = 0; i < 8; ++i) gof[(size_t)(mb + i) * N + col] = acc[i];
  }
}

// ---------------- weight pre-pack: fp32 [K][N] -> bf16 WMMA-B fragment layout ----------------
__global__ void pack_weight(const float* __restrict__ src, __bf16* __restrict__ dst,
                            int K, int N, int transpose) {
  const int total = K * N;
  const int nkc = K >> 5;
  for (int idx = blockIdx.x * blockDim.x + threadIdx.x; idx < total; idx += gridDim.x * blockDim.x) {
    int e = idx & 15;
    int L = (idx >> 4) & 31;
    int rest = idx >> 9;
    int kc = rest % nkc;
    int T  = rest / nkc;
    int n = T * 16 + (L & 15);
    int k = kc * 32 + ((L < 16) ? 0 : 8) + (e & 7) + ((e & 8) ? 16 : 0);
    float v = transpose ? src[(size_t)n * K + k] : src[(size_t)k * N + n];
    dst[idx] = (__bf16)v;
  }
}

// ---------------- persistent scan kernel: one workgroup, all 8 batch rows ----------------
struct ScanArgs {
  const float *x, *sb_init, *sb_bq, *sb_bw, *sb_bo, *p_bin, *bbq, *bp_b;
  const float *dn_bqkv, *dn_wbeta, *dn_bbeta, *dn_bo, *dn_lng, *dn_lnb;
  const float *sup_b, *op_b;
  const __bf16 *pk_sbwq, *pk_sbww, *pk_sbwo, *pk_pwin, *pk_bwq, *pk_bkT, *pk_bpw;
  const __bf16 *pk_qkv0, *pk_qkv1, *pk_wo0, *pk_wo1, *pk_supw, *pk_opw;
  float *slots, *St, *y;
};

#define SMEM_BYTES 184832

__global__ __launch_bounds__(NT, 1) void scan_kernel(ScanArgs A) {
  extern __shared__ char smem[];
  char* p = smem;
  __bf16* tokb = (__bf16*)p; p += 16 * 512 * 2;
  __bf16* xhb  = (__bf16*)p; p += 16 * 256 * 2;
  __bf16* obb  = (__bf16*)p; p += 16 * 256 * 2;
  __bf16* smb  = (__bf16*)p; p += 16 * 128 * 2;
  float* xhf   = (float*)p;  p += 16 * 256 * 4;
  float* qf    = (float*)p;  p += 16 * 128 * 4;
  float* wrf   = (float*)p;  p += 16 * 128 * 4;
  float* ab    = (float*)p;  p += 16 * 64 * 4;
  float* bab   = (float*)p;  p += 16 * 64 * 4;
  float* betab = (float*)p;  p += 16 * 4 * 4;
  float* kscl  = (float*)p;  p += 16 * 4 * 4;
  float* skb   = (float*)p;  p += 8 * 4 * 64 * 4;
  float* qkvf  = (float*)p;  p += 16 * 768 * 4;
  float* gof   = (float*)p;  p += 16 * 768 * 4;

  const int tid = threadIdx.x, lane = tid & 31, wave = tid >> 5;
  const float rscale = 0.08838834764831845f;   // 1/sqrt(128)
  const float4 f40 = make_float4(0.f, 0.f, 0.f, 0.f);

  // ---- init state ----
  for (int i = tid; i < Bv * Nv * SDv; i += NT) A.slots[i] = A.sb_init[i % (Nv * SDv)];
  for (int i = tid; i < 2 * Bv * NHv * DHv * DHv; i += NT) A.St[i] = 0.f;
  for (int i = tid; i < 16 * 256; i += NT) { obb[i] = (__bf16)0.f; xhf[i] = 0.f; xhb[i] = (__bf16)0.f; }
  __syncthreads();

  for (int t = 0; t < Sv; ++t) {
    // stage token (bf16, pad rows zero)
    for (int i4 = tid; i4 < 16 * 128; i4 += NT) {
      int b = i4 >> 7, dd = i4 & 127;
      float4 v = (b < Bv) ? ((const float4*)(A.x + ((size_t)b * Sv + t) * Dv))[dd] : f40;
      store_bf4(&tokb[i4 * 4], v);
    }
    __syncthreads();

    // q = tok @ sb_wq + sb_bq
    gemm_ws(tokb, 512, A.pk_sbwq, 128, gof, wave, lane);
    __syncthreads();
    for (int i4 = tid; i4 < 16 * 32; i4 += NT)
      ((float4*)qf)[i4] = f4add(((const float4*)gof)[i4], ((const float4*)A.sb_bq)[i4 & 31]);
    __syncthreads();

    // wr = tok @ sb_ww + sb_bw ; logits a = (q . slots)/scale
    gemm_ws(tokb, 512, A.pk_sbww, 128, gof, wave, lane);
    __syncthreads();
    for (int i4 = tid; i4 < 16 * 32; i4 += NT)
      ((float4*)wrf)[i4] = f4add(((const float4*)gof)[i4], ((const float4*)A.sb_bw)[i4 & 31]);
    {
      int b = tid >> 6, n = tid & 63;                  // all 512 threads
      const float4* sp = (const float4*)(A.slots + ((size_t)(b * Nv + n)) * SDv);
      const float4* qp = (const float4*)(qf + b * SDv);
      float acc = 0.f;
      for (int d = 0; d < 32; ++d) acc += dot4(qp[d], sp[d]);
      ab[tid] = acc * rscale;
    }
    __syncthreads();
    if (wave < Bv) {                                   // softmax a (wave-parallel)
      float* r = ab + wave * 64;
      float v0 = r[lane], v1 = r[lane + 32];
      float m = wave_max(fmaxf(v0, v1));
      float e0 = __expf(v0 - m), e1 = __expf(v1 - m);
      float inv = 1.f / wave_sum(e0 + e1);
      r[lane] = e0 * inv; r[lane + 32] = e1 * inv;
    }
    __syncthreads();
    // read = sum_n a[n] slots[n] -> smb
    {
      int b = tid >> 5, d4 = tid & 31;
      float4 acc = f40;
      if (b < Bv) {
        const float* ar = ab + b * 64;
        const float4* sp = (const float4*)(A.slots + (size_t)b * Nv * SDv) + d4;
        for (int n = 0; n < Nv; ++n) {
          float an = ar[n]; float4 sv = sp[(size_t)n * 32];
          acc.x += an * sv.x; acc.y += an * sv.y; acc.z += an * sv.z; acc.w += an * sv.w;
        }
      }
      store_bf4(&smb[tid * 4], acc);
    }
    __syncthreads();
    // slot_out = read @ sb_wo + sb_bo -> y ; gated slot update #1
    gemm_ws(smb, 128, A.pk_sbwo, 512, gof, wave, lane);
    __syncthreads();
    for (int i4 = tid; i4 < Bv * 128; i4 += NT) {
      int b = i4 >> 7, dd = i4 & 127;
      float4 v = f4add(((const float4*)(gof + b * 512))[dd], ((const float4*)A.sb_bo)[dd]);
      ((float4*)(A.y + ((size_t)b * Sv + t) * Dv))[dd] = v;
    }
    for (int i4 = tid; i4 < Bv * Nv * 32; i4 += NT) {
      int d4 = i4 & 31, n = (i4 >> 5) & 63, b = i4 >> 11;
      float av = ab[b * 64 + n];
      float g = sigmoidf_(av);
      float4* sp = (float4*)A.slots + i4;
      float4 s = *sp;
      float4 w = ((const float4*)(wrf + b * SDv))[d4];
      s.x = (1.f - g) * s.x + g * (av * w.x);
      s.y = (1.f - g) * s.y + g * (av * w.y);
      s.z = (1.f - g) * s.z + g * (av * w.z);
      s.w = (1.f - g) * s.w + g * (av * w.w);
      *sp = s;
    }
    __syncthreads();

    // xh = tok @ p_win + p_bin
    gemm_ws(tokb, 512, A.pk_pwin, 256, gof, wave, lane);
    __syncthreads();
    for (int i4 = tid; i4 < 16 * 64; i4 += NT) {
      int b = i4 >> 6, dd = i4 & 63;
      float4 v = (b < Bv) ? f4add(((const float4*)gof)[i4], ((const float4*)A.p_bin)[dd]) : f40;
      ((float4*)xhf)[i4] = v;
      store_bf4(&xhb[i4 * 4], v);
    }
    __syncthreads();

    // bq = xh @ b_wq + b_bq -> smb
    gemm_ws(xhb, 256, A.pk_bwq, 128, gof, wave, lane);
    __syncthreads();
    {
      int b = tid >> 5, dd = tid & 31;
      float4 v = (b < Bv) ? f4add(((const float4*)gof)[tid], ((const float4*)A.bbq)[dd]) : f40;
      store_bf4(&smb[tid * 4], v);
    }
    __syncthreads();
    // ba = softmax(bq @ b_keys^T / scale)
    gemm_ws(smb, 128, A.pk_bkT, 64, gof, wave, lane);
    __syncthreads();
    if (wave < Bv) {
      const float* r = gof + wave * 64;
      float* w = bab + wave * 64;
      float v0 = r[lane] * rscale, v1 = r[lane + 32] * rscale;
      float m = wave_max(fmaxf(v0, v1));
      float e0 = __expf(v0 - m), e1 = __expf(v1 - m);
      float inv = 1.f / wave_sum(e0 + e1);
      w[lane] = e0 * inv; w[lane + 32] = e1 * inv;
    }
    __syncthreads();
    // bindings = sum_n ba[n] slots[n] -> smb
    {
      int b = tid >> 5, d4 = tid & 31;
      float4 acc = f40;
      if (b < Bv) {
        const float* br = bab + b * 64;
        const float4* sp = (const float4*)(A.slots + (size_t)b * Nv * SDv) + d4;
        for (int n = 0; n < Nv; ++n) {
          float an = br[n]; float4 sv = sp[(size_t)n * 32];
          acc.x += an * sv.x; acc.y += an * sv.y; acc.z += an * sv.z; acc.w += an * sv.w;
        }
      }
      store_bf4(&smb[tid * 4], acc);
    }
    __syncthreads();
    // xh += bindings @ bp_w + bp_b
    gemm_ws(smb, 128, A.pk_bpw, 256, gof, wave, lane);
    __syncthreads();
    for (int i4 = tid; i4 < 16 * 64; i4 += NT) {
      int b = i4 >> 6, dd = i4 & 63;
      float4 v = f40;
      if (b < Bv)
        v = f4add(f4add(((const float4*)xhf)[i4], ((const float4*)gof)[i4]), ((const float4*)A.bp_b)[dd]);
      ((float4*)xhf)[i4] = v;
      store_bf4(&xhb[i4 * 4], v);
    }
    __syncthreads();

    // ---- DeltaNet stack (2 layers) ----
    for (int l = 0; l < 2; ++l) {
      gemm_ws(xhb, 256, l ? A.pk_qkv1 : A.pk_qkv0, 768, gof, wave, lane);
      __syncthreads();
      const float* bq_ = A.dn_bqkv + l * 768;
      for (int i4 = tid; i4 < 16 * 192; i4 += NT) {
        int b = i4 / 192, dd = i4 - b * 192;
        ((float4*)qkvf)[i4] = f4add(((const float4*)gof)[i4], ((const float4*)bq_)[dd]);
      }
      __syncthreads();
      // beta + k-norm scale per (b,h): 16-lane subgroup reductions
      {
        int combo = tid >> 4, sub = tid & 15;          // combo = b*4+h
        int b = combo >> 2, h = combo & 3;
        const float* xr = xhf + b * 256;
        const float* wb = A.dn_wbeta + l * 1024;
        float acc = 0.f;
        for (int d = sub; d < 256; d += 16) acc += xr[d] * wb[d * 4 + h];
        acc = grp16_sum(acc);
        const float* kp = qkvf + b * 768 + 256 + h * 64;
        float nrm = 0.f;
        for (int j = sub; j < 64; j += 16) nrm += kp[j] * kp[j];
        nrm = grp16_sum(nrm);
        if (sub == 0) {
          betab[combo] = sigmoidf_(acc + A.dn_bbeta[l * 4 + h]);
          kscl[combo]  = 1.f / (sqrtf(nrm) + 1e-6f);
        }
      }
      __syncthreads();
      {
        int b = tid >> 6, dd = tid & 63;               // normalize k (float4)
        float sc = kscl[b * 4 + (dd >> 4)];
        float4* kp = (float4*)(qkvf + b * 768 + 256) + dd;
        float4 kv = *kp;
        kv.x *= sc; kv.y *= sc; kv.z *= sc; kv.w *= sc;
        *kp = kv;
      }
      __syncthreads();
      // Sk = S @ k
      for (int r = tid; r < Bv * NHv * DHv; r += NT) {
        int i2 = r & 63, h = (r >> 6) & 3, b = r >> 8;
        const float4* Sp = (const float4*)(A.St + ((((size_t)l * Bv + b) * NHv + h) * DHv + i2) * DHv);
        const float4* kp = (const float4*)(qkvf + b * 768 + 256 + h * 64);
        float acc = 0.f;
        for (int j = 0; j < 16; ++j) acc += dot4(Sp[j], kp[j]);
        skb[r] = acc;
      }
      __syncthreads();
      // S += beta*(v-Sk) x k ; o = S_new @ q
      for (int r = tid; r < Bv * NHv * DHv; r += NT) {
        int i2 = r & 63, h = (r >> 6) & 3, b = r >> 8;
        float4* Sp = (float4*)(A.St + ((((size_t)l * Bv + b) * NHv + h) * DHv + i2) * DHv);
        const float4* kp = (const float4*)(qkvf + b * 768 + 256 + h * 64);
        const float4* qp = (const float4*)(qkvf + b * 768 + h * 64);
        float u = betab[b * 4 + h] * (qkvf[b * 768 + 512 + h * 64 + i2] - skb[r]);
        float acc = 0.f;
        for (int j = 0; j < 16; ++j) {
          float4 s = Sp[j], kv = kp[j], qv = qp[j];
          s.x += u * kv.x; s.y += u * kv.y; s.z += u * kv.z; s.w += u * kv.w;
          Sp[j] = s;
          acc += dot4(s, qv);
        }
        obb[b * 256 + h * 64 + i2] = (__bf16)acc;
      }
      __syncthreads();
      // xh = LN(xh + o @ dn_wo + dn_bo)  (wave-parallel LN)
      gemm_ws(obb, 256, l ? A.pk_wo1 : A.pk_wo0, 256, gof, wave, lane);
      __syncthreads();
      if (wave < Bv) {
        const float* bo = A.dn_bo + l * Hv;
        const float* gl = A.dn_lng + l * Hv;
        const float* bl = A.dn_lnb + l * Hv;
        float* xr = xhf + wave * Hv;
        const float* gr = gof + wave * Hv;
        float h2v[8];
        float s = 0.f, ss = 0.f;
#pragma unroll
        for (int j = 0; j < 8; ++j) {
          int d = lane + j * 32;
          float h2 = xr[d] + gr[d] + bo[d];
          h2v[j] = h2; s += h2; ss += h2 * h2;
        }
        s = wave_sum(s); ss = wave_sum(ss);
        float mean = s * (1.f / Hv);
        float var = ss * (1.f / Hv) - mean * mean;
        float inv = rsqrtf(var + 1e-5f);
#pragma unroll
        for (int j = 0; j < 8; ++j) {
          int d = lane + j * 32;
          float v = (h2v[j] - mean) * inv * gl[d] + bl[d];
          xr[d] = v; xhb[wave * Hv + d] = (__bf16)v;
        }
      }
      __syncthreads();
    }

    // xw = xh @ sup_w ; gated slot update #2
    gemm_ws(xhb, 256, A.pk_supw, 128, gof, wave, lane);
    __syncthreads();
    for (int i4 = tid; i4 < Bv * Nv * 32; i4 += NT) {
      int d4 = i4 & 31, n = (i4 >> 5) & 63, b = i4 >> 11;
      float bav = bab[b * 64 + n];
      float g2 = sigmoidf_(bav);
      float4 xw = ((const float4*)(gof + b * 128))[d4];
      float4 sb = ((const float4*)A.sup_b)[d4];
      float4* sp = (float4*)A.slots + i4;
      float4 s = *sp;
      s.x = (1.f - g2) * s.x + g2 * (bav * xw.x + sb.x);
      s.y = (1.f - g2) * s.y + g2 * (bav * xw.y + sb.y);
      s.z = (1.f - g2) * s.z + g2 * (bav * xw.z + sb.z);
      s.w = (1.f - g2) * s.w + g2 * (bav * xw.w + sb.w);
      *sp = s;
    }
    __syncthreads();
    // out = slot_out + xh @ op_w + op_b
    gemm_ws(xhb, 256, A.pk_opw, 512, gof, wave, lane);
    __syncthreads();
    for (int i4 = tid; i4 < Bv * 128; i4 += NT) {
      int b = i4 >> 7, dd = i4 & 127;
      float4* yp = (float4*)(A.y + ((size_t)b * Sv + t) * Dv) + dd;
      float4 v = f4add(((const float4*)(gof + b * 512))[dd], ((const float4*)A.op_b)[dd]);
      *yp = f4add(*yp, v);
    }
    __syncthreads();
  }
}

// ---------------- final LN(y) -> bf16 rows ----------------
__global__ void ln_rows(const float* __restrict__ y, const float* __restrict__ g,
                        const float* __restrict__ b, __bf16* __restrict__ out) {
  int row = blockIdx.x * blockDim.x + threadIdx.x;
  if (row >= Bv * Sv) return;
  const float4* r = (const float4*)(y + (size_t)row * Dv);
  float s = 0.f, ss = 0.f;
  for (int d = 0; d < 128; ++d) {
    float4 v = r[d];
    s += v.x + v.y + v.z + v.w;
    ss += dot4(v, v);
  }
  float mean = s * (1.f / Dv);
  float var = ss * (1.f / Dv) - mean * mean;
  float inv = rsqrtf(var + 1e-5f);
  __bf16* o = out + (size_t)row * Dv;
  const float4* g4 = (const float4*)g;
  const float4* b4 = (const float4*)b;
  for (int d = 0; d < 128; ++d) {
    float4 v = r[d], gg = g4[d], bb = b4[d];
    float4 w = make_float4((v.x - mean) * inv * gg.x + bb.x, (v.y - mean) * inv * gg.y + bb.y,
                           (v.z - mean) * inv * gg.z + bb.z, (v.w - mean) * inv * gg.w + bb.w);
    store_bf4(&o[d * 4], w);
  }
}

// ---------------- final projection: (8192x512) @ (512x512), 16x32 tile per wave ----------------
__global__ __launch_bounds__(256) void out_gemm(const __bf16* __restrict__ aln,
                                                const __bf16* __restrict__ pack,
                                                const float* __restrict__ bias,
                                                float* __restrict__ out) {
  const int lane = threadIdx.x & 31, wave = threadIdx.x >> 5;
  const int job = blockIdx.x * 8 + wave;      // 8192 jobs: 512 tm x 16 tn-pairs
  const int tm = job >> 4, tn0 = (job & 15) * 2;
  if (tm >= (Bv * Sv) / 16) return;
  const int K = Dv, nkc = K >> 5;
  v8f acc0 = {}, acc1 = {};
  const __bf16* arow = aln + (size_t)(tm * 16 + (lane & 15)) * K + ((lane & 16) ? 8 : 0);
  for (int kc = 0; kc < nkc; ++kc) {
    v16bf a;
    const __bf16* ap = arow + kc * 32;
#pragma unroll
    for (int i = 0; i < 8; ++i) { a[i] = ap[i]; a[i + 8] = ap[16 + i]; }
    v16bf b0 = load_b_frag(pack, nkc, tn0, kc, lane);
    v16bf b1 = load_b_frag(pack, nkc, tn0 + 1, kc, lane);
    acc0 = __builtin_amdgcn_wmma_f32_16x16x32_bf16(false, a, false, b0, (short)0, acc0, false, false);
    acc1 = __builtin_amdgcn_wmma_f32_16x16x32_bf16(false, a, false, b1, (short)0, acc1, false, false);
  }
  const int col0 = tn0 * 16 + (lane & 15);
  const int mb   = tm * 16 + ((lane & 16) ? 8 : 0);
#pragma unroll
  for (int i = 0; i < 8; ++i) {
    out[(size_t)(mb + i) * Dv + col0]      = acc0[i] + bias[col0];
    out[(size_t)(mb + i) * Dv + col0 + 16] = acc1[i] + bias[col0 + 16];
  }
}

// ---------------- host ----------------
extern "C" void kernel_launch(void* const* d_in, const int* in_sizes, int n_in,
                              void* d_out, int out_size, void* d_ws, size_t ws_size,
                              hipStream_t stream) {
  (void)in_sizes; (void)n_in; (void)out_size; (void)ws_size;
  const float* x       = (const float*)d_in[0];
  const float* sb_init = (const float*)d_in[1];
  const float* sb_wq   = (const float*)d_in[2];
  const float* sb_bq   = (const float*)d_in[3];
  const float* sb_ww   = (const float*)d_in[4];
  const float* sb_bw   = (const float*)d_in[5];
  const float* sb_wo   = (const float*)d_in[6];
  const float* sb_bo   = (const float*)d_in[7];
  const float* p_win   = (const float*)d_in[8];
  const float* p_bin   = (const float*)d_in[9];
  const float* b_wq    = (const float*)d_in[10];
  const float* b_bq    = (const float*)d_in[11];
  const float* b_keys  = (const float*)d_in[12];
  const float* bp_w    = (const float*)d_in[13];
  const float* bp_b    = (const float*)d_in[14];
  const float* dn_wqkv = (const float*)d_in[15];
  const float* dn_bqkv = (const float*)d_in[16];
  const float* dn_wbeta= (const float*)d_in[17];
  const float* dn_bbeta= (const float*)d_in[18];
  const float* dn_wo   = (const float*)d_in[19];
  const float* dn_bo   = (const float*)d_in[20];
  const float* dn_lng  = (const float*)d_in[21];
  const float* dn_lnb  = (const float*)d_in[22];
  const float* sup_w   = (const float*)d_in[23];
  const float* sup_b   = (const float*)d_in[24];
  const float* op_w    = (const float*)d_in[25];
  const float* op_b    = (const float*)d_in[26];
  const float* oln_g   = (const float*)d_in[27];
  const float* oln_b   = (const float*)d_in[28];
  const float* ow      = (const float*)d_in[29];
  const float* ob      = (const float*)d_in[30];

  char* ws = (char*)d_ws;
  size_t off = 0;
  auto take = [&](size_t bytes) -> char* {
    char* p = ws + off;
    off += (bytes + 255) & ~(size_t)255;
    return p;
  };
  __bf16* pk_sbwq = (__bf16*)take(512 * 128 * 2);
  __bf16* pk_sbww = (__bf16*)take(512 * 128 * 2);
  __bf16* pk_sbwo = (__bf16*)take(128 * 512 * 2);
  __bf16* pk_pwin = (__bf16*)take(512 * 256 * 2);
  __bf16* pk_bwq  = (__bf16*)take(256 * 128 * 2);
  __bf16* pk_bkT  = (__bf16*)take(128 * 64 * 2);
  __bf16* pk_bpw  = (__bf16*)take(128 * 256 * 2);
  __bf16* pk_qkv0 = (__bf16*)take(256 * 768 * 2);
  __bf16* pk_qkv1 = (__bf16*)take(256 * 768 * 2);
  __bf16* pk_wo0  = (__bf16*)take(256 * 256 * 2);
  __bf16* pk_wo1  = (__bf16*)take(256 * 256 * 2);
  __bf16* pk_supw = (__bf16*)take(256 * 128 * 2);
  __bf16* pk_opw  = (__bf16*)take(256 * 512 * 2);
  __bf16* pk_ow   = (__bf16*)take(512 * 512 * 2);
  float* slots = (float*)take((size_t)Bv * Nv * SDv * 4);
  float* St    = (float*)take((size_t)2 * Bv * NHv * DHv * DHv * 4);
  __bf16* ylnb = (__bf16*)take((size_t)Bv * Sv * Dv * 2);

  dim3 pb(256), pg(128);
  pack_weight<<<pg, pb, 0, stream>>>(sb_wq,  pk_sbwq, 512, 128, 0);
  pack_weight<<<pg, pb, 0, stream>>>(sb_ww,  pk_sbww, 512, 128, 0);
  pack_weight<<<pg, pb, 0, stream>>>(sb_wo,  pk_sbwo, 128, 512, 0);
  pack_weight<<<pg, pb, 0, stream>>>(p_win,  pk_pwin, 512, 256, 0);
  pack_weight<<<pg, pb, 0, stream>>>(b_wq,   pk_bwq,  256, 128, 0);
  pack_weight<<<pg, pb, 0, stream>>>(b_keys, pk_bkT,  128, 64, 1);
  pack_weight<<<pg, pb, 0, stream>>>(bp_w,   pk_bpw,  128, 256, 0);
  pack_weight<<<pg, pb, 0, stream>>>(dn_wqkv,             pk_qkv0, 256, 768, 0);
  pack_weight<<<pg, pb, 0, stream>>>(dn_wqkv + 256 * 768, pk_qkv1, 256, 768, 0);
  pack_weight<<<pg, pb, 0, stream>>>(dn_wo,               pk_wo0,  256, 256, 0);
  pack_weight<<<pg, pb, 0, stream>>>(dn_wo + 256 * 256,   pk_wo1,  256, 256, 0);
  pack_weight<<<pg, pb, 0, stream>>>(sup_w,  pk_supw, 256, 128, 0);
  pack_weight<<<pg, pb, 0, stream>>>(op_w,   pk_opw,  256, 512, 0);
  pack_weight<<<pg, pb, 0, stream>>>(ow,     pk_ow,   512, 512, 0);

  ScanArgs a;
  a.x = x; a.sb_init = sb_init; a.sb_bq = sb_bq; a.sb_bw = sb_bw; a.sb_bo = sb_bo;
  a.p_bin = p_bin; a.bbq = b_bq; a.bp_b = bp_b;
  a.dn_bqkv = dn_bqkv; a.dn_wbeta = dn_wbeta; a.dn_bbeta = dn_bbeta; a.dn_bo = dn_bo;
  a.dn_lng = dn_lng; a.dn_lnb = dn_lnb; a.sup_b = sup_b; a.op_b = op_b;
  a.pk_sbwq = pk_sbwq; a.pk_sbww = pk_sbww; a.pk_sbwo = pk_sbwo; a.pk_pwin = pk_pwin;
  a.pk_bwq = pk_bwq; a.pk_bkT = pk_bkT; a.pk_bpw = pk_bpw;
  a.pk_qkv0 = pk_qkv0; a.pk_qkv1 = pk_qkv1; a.pk_wo0 = pk_wo0; a.pk_wo1 = pk_wo1;
  a.pk_supw = pk_supw; a.pk_opw = pk_opw;
  a.slots = slots; a.St = St; a.y = (float*)d_out;

  scan_kernel<<<dim3(1), dim3(NT), SMEM_BYTES, stream>>>(a);

  ln_rows<<<dim3((Bv * Sv + 127) / 128), dim3(128), 0, stream>>>((const float*)d_out, oln_g, oln_b, ylnb);
  out_gemm<<<dim3(((Bv * Sv) / 16) * (Dv / 16) / 16), dim3(256), 0, stream>>>(ylnb, pk_ow, ob, (float*)d_out);
}
